// SoftTCMLayer_3959959847547
// MI455X (gfx1250) — matrix operand order repaired and checked by reading, not compiled, not run on hardware
//
#include <hip/hip_runtime.h>

// ---------------- problem constants ----------------
#define BB   4
#define TT   1024
#define CC   1024
#define HH   16
#define HD   64
#define DD   128          // DI
#define BT   4096         // BB*TT
#define C4   4096         // 4*CC

typedef __attribute__((ext_vector_type(16))) __bf16 v16bf;
typedef __attribute__((ext_vector_type(8)))  __bf16 v8bf;
typedef __attribute__((ext_vector_type(8)))  float  v8f;

#define CAT16(h0, h1) __builtin_shufflevector(h0, h1, 0,1,2,3,4,5,6,7,8,9,10,11,12,13,14,15)

// immediate-offset xor swizzle within 32 lanes (and_mask=0x1f)
template <int XMASK>
__device__ __forceinline__ float swz_max(float x) {
    return fmaxf(x, __int_as_float(__builtin_amdgcn_ds_swizzle(
                        __float_as_int(x), (XMASK << 10) | 0x1f)));
}
template <int XMASK>
__device__ __forceinline__ float swz_add(float x) {
    return x + __int_as_float(__builtin_amdgcn_ds_swizzle(
                   __float_as_int(x), (XMASK << 10) | 0x1f));
}

// =====================================================================
// Weight convert + transpose:  W[K,N] f32  ->  Wt[N,K] bf16
// =====================================================================
__global__ void transpose_w_kernel(const float* __restrict__ W,
                                   __bf16* __restrict__ out, int K, int N) {
    long long idx = (long long)blockIdx.x * blockDim.x + threadIdx.x;
    if (idx >= (long long)K * N) return;
    int k = (int)(idx / N);
    int n = (int)(idx % N);
    out[(long long)n * K + k] = (__bf16)W[idx];
}

// v_bf [B,T,C]=[B,T,H,hd] bf16 -> vt [B,H,hd,T] bf16
__global__ void transpose_v_kernel(const __bf16* __restrict__ vb,
                                   __bf16* __restrict__ vt) {
    long long idx = (long long)blockIdx.x * blockDim.x + threadIdx.x;
    int c = (int)(idx & (CC - 1));
    long long bt = idx >> 10;
    int b = (int)(bt / TT);
    int t = (int)(bt % TT);
    int h = c >> 6;
    int d = c & 63;
    vt[(((long long)(b * HH + h)) * HD + d) * TT + t] = vb[idx];
}

// =====================================================================
// LayerNorm: one wave per token
// =====================================================================
__global__ __launch_bounds__(256) void layernorm_kernel(
    const float* __restrict__ x, const float* __restrict__ w,
    const float* __restrict__ b, float* __restrict__ outF,
    __bf16* __restrict__ outB) {
    int tok = blockIdx.x * 8 + (threadIdx.x >> 5);
    int l = threadIdx.x & 31;
    const float* xr = x + (long long)tok * CC;
    float vals[32];
    float s = 0.f, s2 = 0.f;
#pragma unroll
    for (int i = 0; i < 32; ++i) {
        float v = xr[l + i * 32];
        vals[i] = v; s += v; s2 += v * v;
    }
    s = swz_add<16>(swz_add<8>(swz_add<4>(swz_add<2>(swz_add<1>(s)))));
    s2 = swz_add<16>(swz_add<8>(swz_add<4>(swz_add<2>(swz_add<1>(s2)))));
    float mu  = s * (1.0f / CC);
    float var = s2 * (1.0f / CC) - mu * mu;
    float rstd = rsqrtf(var + 1e-5f);
#pragma unroll
    for (int i = 0; i < 32; ++i) {
        int c = l + i * 32;
        float y = (vals[i] - mu) * rstd * w[c] + b[c];
        if (outF) outF[(long long)tok * CC + c] = y;
        if (outB) outB[(long long)tok * CC + c] = (__bf16)y;
    }
}

// =====================================================================
// Gearbox: per-token scalars mass / eps / gamma / gate
// =====================================================================
__global__ __launch_bounds__(256) void gearbox_kernel(
    const float* __restrict__ h, const float* __restrict__ hn,
    const float* __restrict__ gw, const float* __restrict__ gb,
    const float* __restrict__ mw, const float* __restrict__ mb,
    const float* __restrict__ ew, const float* __restrict__ eb,
    float* __restrict__ gammaA, float* __restrict__ epsA,
    float* __restrict__ massA, float* __restrict__ gateA) {
    int tok = blockIdx.x * 8 + (threadIdx.x >> 5);
    int l = threadIdx.x & 31;
    const float* hr = h  + (long long)tok * CC;
    const float* nr = hn + (long long)tok * CC;
    float sm = 0.f, se = 0.f, sg = 0.f;
#pragma unroll
    for (int i = 0; i < 32; ++i) {
        int c = l + i * 32;
        float hv = hr[c];
        sm += hv * mw[c];
        se += hv * ew[c];
        sg += nr[c] * gw[c];
    }
    sm = swz_add<16>(swz_add<8>(swz_add<4>(swz_add<2>(swz_add<1>(sm)))));
    se = swz_add<16>(swz_add<8>(swz_add<4>(swz_add<2>(swz_add<1>(se)))));
    sg = swz_add<16>(swz_add<8>(swz_add<4>(swz_add<2>(swz_add<1>(sg)))));
    if (l == 0) {
        float mv = sm + mb[0];
        float sp = (mv > 20.f) ? mv : log1pf(__expf(mv));
        float ms = 1.f + sp;
        massA[tok]  = ms;
        epsA[tok]   = 1.f / (1.f + __expf(-(se + eb[0])));
        gammaA[tok] = 1.f / (1.f + __expf(-(sg + gb[0])));
        gateA[tok]  = 1.f / (1.f + __expf(-(ms - 1.0f)));   // SYS2_THR = 1.0
    }
}

// =====================================================================
// FSI
// =====================================================================
__global__ __launch_bounds__(256) void fsi_kernel(
    const float* __restrict__ hn, const float* __restrict__ fp,
    float* __restrict__ fsi) {
    int tok = blockIdx.x * 8 + (threadIdx.x >> 5);
    int l = threadIdx.x & 31;
    const float* nr = hn + (long long)tok * CC;
    const float* fr = fp + (long long)tok * DD;
    float sh = 0.f, sf = 0.f;
#pragma unroll
    for (int i = 0; i < 32; ++i) { float v = nr[l + i * 32]; sh += v * v; }
#pragma unroll
    for (int i = 0; i < 4;  ++i) { float v = fr[l + i * 32]; sf += v * v; }
    sh = swz_add<16>(swz_add<8>(swz_add<4>(swz_add<2>(swz_add<1>(sh)))));
    sf = swz_add<16>(swz_add<8>(swz_add<4>(swz_add<2>(swz_add<1>(sf)))));
    if (l == 0) fsi[tok] = sqrtf(sh) / (2.f * sqrtf(sf) + 1e-6f);
}

// =====================================================================
// Momentum scan: one thread per (b, di) chain
// =====================================================================
__global__ void scan_kernel(const float* __restrict__ fp,
                            const float* __restrict__ gammaA,
                            const float* __restrict__ epsA,
                            const float* __restrict__ massA,
                            float* __restrict__ m_out,
                            __bf16* __restrict__ m_bf) {
    int tid = blockIdx.x * blockDim.x + threadIdx.x;   // 0..511
    int b  = tid >> 7;
    int di = tid & 127;
    float m = 0.f;
    for (int t = 0; t < TT; ++t) {
        long long row = (long long)b * TT + t;
        float f = fp[row * DD + di];
        m = gammaA[row] * m + (epsA[row] / massA[row]) * f;
        m_out[row * DD + di] = m;
        m_bf[row * DD + di] = (__bf16)m;
    }
}

// =====================================================================
// WMMA bf16 GEMM with LDS tile staging.
// block = 256 thr = 8 waves; block tile 128x128, wave tile 32x64, K step 32.
// MODE: 0 none | 1 exact GELU | 2 acc+bias-0.1*aux | 3 aux+rowscale*(acc+bias)
// =====================================================================
template <int MODE>
__global__ __launch_bounds__(256) void gemm_bf16(
    const __bf16* __restrict__ A, const __bf16* __restrict__ Bt,
    int M, int N, int K,
    const float* __restrict__ bias, const float* __restrict__ aux,
    const float* __restrict__ rowscale,
    float* __restrict__ outF, __bf16* __restrict__ outB) {
    __shared__ __bf16 ldsA[128 * 32];
    __shared__ __bf16 ldsB[128 * 32];

    const int tid = threadIdx.x;
    const int wave = tid >> 5, wm = wave & 3, wn = wave >> 2;
    const int l = tid & 31, l15 = l & 15, lhi = l >> 4;
    const int mblk = blockIdx.y * 128;
    const int nblk = blockIdx.x * 128;

    v8f acc[2][4];
#pragma unroll
    for (int i = 0; i < 2; ++i)
#pragma unroll
        for (int j = 0; j < 4; ++j)
#pragma unroll
            for (int r = 0; r < 8; ++r) acc[i][j][r] = 0.f;

    // cooperative tile-load mapping: 128 rows x 4 chunks(16B) = 512 chunks,
    // 2 per thread
    const int c0 = tid * 2, c1 = tid * 2 + 1;
    const int ar0 = c0 >> 2, ao0 = (c0 & 3) * 8;
    const int ar1 = c1 >> 2, ao1 = (c1 & 3) * 8;
    const __bf16* Ab = A + (long long)mblk * K;
    const __bf16* Bb = Bt + (long long)nblk * K;

    for (int k0 = 0; k0 < K; k0 += 32) {
        __syncthreads();
        *(v8bf*)&ldsA[ar0 * 32 + ao0] = *(const v8bf*)(Ab + (long long)ar0 * K + k0 + ao0);
        *(v8bf*)&ldsA[ar1 * 32 + ao1] = *(const v8bf*)(Ab + (long long)ar1 * K + k0 + ao1);
        *(v8bf*)&ldsB[ar0 * 32 + ao0] = *(const v8bf*)(Bb + (long long)ar0 * K + k0 + ao0);
        *(v8bf*)&ldsB[ar1 * 32 + ao1] = *(const v8bf*)(Bb + (long long)ar1 * K + k0 + ao1);
        if (k0 + 32 < K) {   // warm next K tile into cache
            __builtin_prefetch(Ab + (long long)ar0 * K + k0 + 32 + ao0, 0, 0);
            __builtin_prefetch(Bb + (long long)ar0 * K + k0 + 32 + ao0, 0, 0);
        }
        __syncthreads();

        v16bf af[2];
#pragma unroll
        for (int i = 0; i < 2; ++i) {
            const __bf16* p = &ldsA[(wm * 32 + i * 16 + l15) * 32 + lhi * 8];
            v8bf h0 = *(const v8bf*)p;
            v8bf h1 = *(const v8bf*)(p + 16);
            af[i] = CAT16(h0, h1);
        }
        v16bf bf[4];
#pragma unroll
        for (int j = 0; j < 4; ++j) {
            const __bf16* p = &ldsB[(wn * 64 + j * 16 + l15) * 32 + lhi * 8];
            v8bf h0 = *(const v8bf*)p;
            v8bf h1 = *(const v8bf*)(p + 16);
            bf[j] = CAT16(h0, h1);
        }
#pragma unroll
        for (int j = 0; j < 4; ++j)
#pragma unroll
            for (int i = 0; i < 2; ++i)
                acc[i][j] = __builtin_amdgcn_wmma_f32_16x16x32_bf16(
                    false, af[i], false, bf[j], (short)0, acc[i][j], false, false);
    }

    const int m0 = mblk + wm * 32;
    const int n0 = nblk + wn * 64;
#pragma unroll
    for (int i = 0; i < 2; ++i)
#pragma unroll
        for (int j = 0; j < 4; ++j) {
            int gn = n0 + j * 16 + l15;
            float bsv = bias ? bias[gn] : 0.f;
#pragma unroll
            for (int r = 0; r < 8; ++r) {
                int gm = m0 + i * 16 + r + lhi * 8;
                long long idx = (long long)gm * N + gn;
                float v = acc[i][j][r] + bsv;
                if (MODE == 1) {
                    v = 0.5f * v * (1.f + erff(v * 0.70710678f));
                } else if (MODE == 2) {
                    v = v - 0.1f * aux[idx];           // LAMBDA_CONF
                } else if (MODE == 3) {
                    float rs = rowscale ? rowscale[gm] : 1.f;
                    v = aux[idx] + rs * v;
                }
                if (outF) outF[idx] = v;
                if (outB) outB[idx] = (__bf16)v;
            }
        }
}

// =====================================================================
// Flash attention with WMMA + LDS-staged K/V tiles (shared by 8 waves).
// grid = (T/128, B*H); each wave: 16 query rows x hd=64, online softmax.
// =====================================================================
__global__ __launch_bounds__(256) void flash_attn_kernel(
    const __bf16* __restrict__ qb, const __bf16* __restrict__ kb,
    const __bf16* __restrict__ vt, __bf16* __restrict__ outb) {
    __shared__ __bf16 kT[32 * 64];       // [key 32][d 64]
    __shared__ __bf16 vTl[64 * 32];      // [d 64][key 32]
    __shared__ __bf16 pT[8][16 * 32];    // per-wave P staging

    const int tid = threadIdx.x;
    const int bh = blockIdx.y;
    const int b = bh >> 4;
    const int h = bh & 15;
    const int wave = tid >> 5;
    const int l = tid & 31, l15 = l & 15, lhi = l >> 4;
    const int qrow0 = blockIdx.x * 128 + wave * 16;

    const __bf16* qbase = qb + ((long long)b * TT) * CC + h * HD;
    const __bf16* kbase = kb + ((long long)b * TT) * CC + h * HD;
    const __bf16* vbase = vt + ((long long)bh * HD) * TT;

    // Q fragments, kept in registers for all 1024 keys
    v16bf qa[2];
    {
        const __bf16* qr = qbase + (long long)(qrow0 + l15) * CC;
#pragma unroll
        for (int kk = 0; kk < 2; ++kk) {
            const __bf16* p = qr + kk * 32 + lhi * 8;
            v8bf h0 = *(const v8bf*)p;
            v8bf h1 = *(const v8bf*)(p + 16);
            qa[kk] = CAT16(h0, h1);
        }
    }

    v8f acc[4];
#pragma unroll
    for (int j = 0; j < 4; ++j)
#pragma unroll
        for (int r = 0; r < 8; ++r) acc[j][r] = 0.f;
    float mrow[8], lrow[8];
#pragma unroll
    for (int r = 0; r < 8; ++r) { mrow[r] = -1e30f; lrow[r] = 0.f; }

    const int krow = tid >> 3, koff = (tid & 7) * 8;   // 32 x 8 chunks
    const int vrow = tid >> 2, voff = (tid & 3) * 8;   // 64 x 4 chunks

    for (int j0 = 0; j0 < TT; j0 += 32) {
        __syncthreads();
        *(v8bf*)&kT[krow * 64 + koff] =
            *(const v8bf*)(kbase + (long long)(j0 + krow) * CC + koff);
        *(v8bf*)&vTl[vrow * 32 + voff] =
            *(const v8bf*)(vbase + (long long)vrow * TT + j0 + voff);
        __syncthreads();

        // ---- S = Q K^T ----
        v8f s[2];
#pragma unroll
        for (int jt = 0; jt < 2; ++jt) {
#pragma unroll
            for (int r = 0; r < 8; ++r) s[jt][r] = 0.f;
#pragma unroll
            for (int kk = 0; kk < 2; ++kk) {
                const __bf16* p = &kT[(jt * 16 + l15) * 64 + kk * 32 + lhi * 8];
                v8bf h0 = *(const v8bf*)p;
                v8bf h1 = *(const v8bf*)(p + 16);
                s[jt] = __builtin_amdgcn_wmma_f32_16x16x32_bf16(
                    false, qa[kk], false, CAT16(h0, h1), (short)0, s[jt], false, false);
            }
        }

        // ---- online softmax (half-wave row reductions via ds_swizzle) ----
        const float scale = 0.125f;   // 1/sqrt(64)/TAU
        float p0v[8], p1v[8];
#pragma unroll
        for (int r = 0; r < 8; ++r) {
            float a0 = s[0][r] * scale, a1 = s[1][r] * scale;
            float mx = fmaxf(a0, a1);
            mx = swz_max<8>(swz_max<4>(swz_max<2>(swz_max<1>(mx))));
            float mnew = fmaxf(mrow[r], mx);
            float alpha = __expf(mrow[r] - mnew);
            float e0 = __expf(a0 - mnew), e1 = __expf(a1 - mnew);
            float rs = e0 + e1;
            rs = swz_add<8>(swz_add<4>(swz_add<2>(swz_add<1>(rs))));
            lrow[r] = lrow[r] * alpha + rs;
            mrow[r] = mnew;
#pragma unroll
            for (int j = 0; j < 4; ++j) acc[j][r] *= alpha;
            p0v[r] = e0; p1v[r] = e1;
        }

        // ---- stage P: C-layout -> A-fragment via per-wave LDS ----
        __bf16* pw = &pT[wave][0];
#pragma unroll
        for (int r = 0; r < 8; ++r) {
            int M = r + lhi * 8;
            pw[M * 32 + l15]      = (__bf16)p0v[r];
            pw[M * 32 + 16 + l15] = (__bf16)p1v[r];
        }
        v16bf pa;
        {
            const __bf16* src = &pw[l15 * 32 + lhi * 8];
            v8bf h0 = *(const v8bf*)src;
            v8bf h1 = *(const v8bf*)(src + 16);
            pa = CAT16(h0, h1);
        }

        // ---- O += P @ V ----
#pragma unroll
        for (int dt = 0; dt < 4; ++dt) {
            const __bf16* p = &vTl[(dt * 16 + l15) * 32 + lhi * 8];
            v8bf h0 = *(const v8bf*)p;
            v8bf h1 = *(const v8bf*)(p + 16);
            acc[dt] = __builtin_amdgcn_wmma_f32_16x16x32_bf16(
                false, pa, false, CAT16(h0, h1), (short)0, acc[dt], false, false);
        }
    }

    // ---- normalize, store force [B,T,H,hd] bf16 ----
#pragma unroll
    for (int dt = 0; dt < 4; ++dt)
#pragma unroll
        for (int r = 0; r < 8; ++r) {
            float v = acc[dt][r] / lrow[r];
            int M = qrow0 + r + lhi * 8;
            int n = dt * 16 + l15;
            outb[((long long)(b * TT + M)) * CC + h * HD + n] = (__bf16)v;
        }
}

// =====================================================================
// host launch
// =====================================================================
extern "C" void kernel_launch(void* const* d_in, const int* in_sizes, int n_in,
                              void* d_out, int out_size, void* d_ws, size_t ws_size,
                              hipStream_t stream) {
    const float* h      = (const float*)d_in[0];
    const float* ln_w   = (const float*)d_in[1];
    const float* ln_b   = (const float*)d_in[2];
    const float* w_q    = (const float*)d_in[3];
    const float* w_k    = (const float*)d_in[4];
    const float* w_v    = (const float*)d_in[5];
    const float* w_o    = (const float*)d_in[6];
    const float* b_o    = (const float*)d_in[7];
    const float* U      = (const float*)d_in[8];
    const float* V      = (const float*)d_in[9];
    const float* gamma_w = (const float*)d_in[10];
    const float* gamma_b = (const float*)d_in[11];
    const float* mass_w  = (const float*)d_in[12];
    const float* mass_b  = (const float*)d_in[13];
    const float* eps_w   = (const float*)d_in[14];
    const float* eps_b   = (const float*)d_in[15];
    const float* lg_w1  = (const float*)d_in[16];
    const float* lg_b1  = (const float*)d_in[17];
    const float* lg_w2  = (const float*)d_in[18];
    const float* lg_b2  = (const float*)d_in[19];
    const float* mlp_w1 = (const float*)d_in[20];
    const float* mlp_b1 = (const float*)d_in[21];
    const float* mlp_w2 = (const float*)d_in[22];
    const float* mlp_b2 = (const float*)d_in[23];

    char* ws = (char*)d_ws;
    const size_t MB = 1024 * 1024;
    float*  hnF    = (float*) (ws + 0 * MB);     // 16 MB
    __bf16* hnB    = (__bf16*)(ws + 16 * MB);    //  8 MB
    __bf16* wqT    = (__bf16*)(ws + 24 * MB);
    __bf16* wkT    = (__bf16*)(ws + 26 * MB);
    __bf16* wvT    = (__bf16*)(ws + 28 * MB);
    __bf16* woT    = (__bf16*)(ws + 30 * MB);
    __bf16* UT     = (__bf16*)(ws + 32 * MB);
    __bf16* VT     = (__bf16*)(ws + 32 * MB + 512 * 1024);
    __bf16* lg1T   = (__bf16*)(ws + 33 * MB);
    __bf16* lg2T   = (__bf16*)(ws + 41 * MB);
    __bf16* mlp1T  = (__bf16*)(ws + 49 * MB);
    __bf16* mlp2T  = (__bf16*)(ws + 57 * MB);
    __bf16* qB     = (__bf16*)(ws + 65 * MB);
    __bf16* kB     = (__bf16*)(ws + 73 * MB);
    __bf16* vB     = (__bf16*)(ws + 81 * MB);
    __bf16* vtB    = (__bf16*)(ws + 89 * MB);
    __bf16* forceB = (__bf16*)(ws + 97 * MB);
    __bf16* ftotB  = (__bf16*)(ws + 105 * MB);
    float*  fproj  = (float*) (ws + 113 * MB);   // 2 MB
    float*  gammaA = (float*) (ws + 115 * MB);
    float*  epsA   = gammaA + BT;
    float*  massA  = gammaA + 2 * BT;
    float*  gateA  = gammaA + 3 * BT;
    __bf16* mB     = (__bf16*)(ws + 116 * MB);   // 1 MB
    float*  hnewF  = (float*) (ws + 117 * MB);   // 16 MB
    __bf16* hn2B   = (__bf16*)(ws + 133 * MB);   // 8 MB (reused for hn3)
    __bf16* t1B    = (__bf16*)(ws + 141 * MB);   // 32 MB (reused for t2)

    float* out_h   = (float*)d_out;
    float* out_m   = out_h + (long long)BT * CC;
    float* out_fsi = out_m + (long long)BT * DD;

    dim3 blk(256);
    auto cvt = [&](const float* W, __bf16* dst, int K, int N) {
        int total = K * N;
        transpose_w_kernel<<<(total + 255) / 256, blk, 0, stream>>>(W, dst, K, N);
    };
    cvt(w_q, wqT, CC, CC);    cvt(w_k, wkT, CC, CC);
    cvt(w_v, wvT, CC, CC);    cvt(w_o, woT, CC, CC);
    cvt(U, UT, CC, DD);       cvt(V, VT, DD, CC);
    cvt(lg_w1, lg1T, CC, C4); cvt(lg_w2, lg2T, C4, CC);
    cvt(mlp_w1, mlp1T, CC, C4); cvt(mlp_w2, mlp2T, C4, CC);

    layernorm_kernel<<<BT / 8, blk, 0, stream>>>(h, ln_w, ln_b, hnF, hnB);
    gearbox_kernel<<<BT / 8, blk, 0, stream>>>(h, hnF, gamma_w, gamma_b,
                                               mass_w, mass_b, eps_w, eps_b,
                                               gammaA, epsA, massA, gateA);
    dim3 gC(CC / 128, BT / 128);
    gemm_bf16<0><<<gC, blk, 0, stream>>>(hnB, wqT, BT, CC, CC, nullptr, nullptr, nullptr, nullptr, qB);
    gemm_bf16<0><<<gC, blk, 0, stream>>>(hnB, wkT, BT, CC, CC, nullptr, nullptr, nullptr, nullptr, kB);
    gemm_bf16<0><<<gC, blk, 0, stream>>>(hnB, wvT, BT, CC, CC, nullptr, nullptr, nullptr, nullptr, vB);
    transpose_v_kernel<<<(BT * CC) / 256, blk, 0, stream>>>(vB, vtB);
    flash_attn_kernel<<<dim3(TT / 128, BB * HH), blk, 0, stream>>>(qB, kB, vtB, forceB);
    // f_total = force@w_o + b_o - 0.1*h_norm
    gemm_bf16<2><<<gC, blk, 0, stream>>>(forceB, woT, BT, CC, CC, b_o, hnF, nullptr, nullptr, ftotB);
    // f_proj = f_total @ U
    gemm_bf16<0><<<dim3(1, BT / 128), blk, 0, stream>>>(ftotB, UT, BT, DD, CC, nullptr, nullptr, nullptr, fproj, nullptr);
    fsi_kernel<<<BT / 8, blk, 0, stream>>>(hnF, fproj, out_fsi);
    scan_kernel<<<2, blk, 0, stream>>>(fproj, gammaA, epsA, massA, out_m, mB);
    // h_new = h + eps * (m @ V)
    gemm_bf16<3><<<gC, blk, 0, stream>>>(mB, VT, BT, CC, DD, nullptr, h, epsA, hnewF, nullptr);
    layernorm_kernel<<<BT / 8, blk, 0, stream>>>(hnewF, ln_w, ln_b, nullptr, hn2B);
    dim3 g4(C4 / 128, BT / 128);
    gemm_bf16<1><<<g4, blk, 0, stream>>>(hn2B, lg1T, BT, C4, CC, lg_b1, nullptr, nullptr, nullptr, t1B);
    gemm_bf16<3><<<gC, blk, 0, stream>>>(t1B, lg2T, BT, CC, C4, lg_b2, hnewF, gateA, out_h, nullptr);
    layernorm_kernel<<<BT / 8, blk, 0, stream>>>(out_h, ln_w, ln_b, nullptr, hn2B);
    gemm_bf16<1><<<g4, blk, 0, stream>>>(hn2B, mlp1T, BT, C4, CC, mlp_b1, nullptr, nullptr, nullptr, t1B);
    gemm_bf16<3><<<gC, blk, 0, stream>>>(t1B, mlp2T, BT, CC, C4, mlp_b2, out_h, nullptr, out_h, nullptr);
}
